// BiRNN_46961172414596
// MI455X (gfx1250) — matrix operand order repaired and checked by reading, not compiled, not run on hardware
//
#include <hip/hip_runtime.h>
#include <hip/hip_bf16.h>
#include <math.h>

// Problem constants (match reference)
#define SEQ_LEN 2048
#define BATCH   128
#define IN_DIM  64
#define HID     128
#define ROWS    (SEQ_LEN * BATCH)   // 262144 (t,b) rows

typedef __attribute__((ext_vector_type(16))) _Float16 v16h;
typedef __attribute__((ext_vector_type(8)))  float    v8f;
typedef unsigned int u32x4 __attribute__((ext_vector_type(4)));
typedef int          i32x8 __attribute__((ext_vector_type(8)));
typedef int          i32x4 __attribute__((ext_vector_type(4)));

union V16H { v16h v; _Float16 h[16]; float4 q[2]; };
union V8F  { v8f  v; float    f[8]; };

// ---- workspace layout (bytes) ----
static constexpr size_t OFF_WXT   = 0;                                   // f16 [64][128]   16 KB
static constexpr size_t OFF_WHT   = 16384;                               // f16 [128][128]  32 KB
static constexpr size_t OFF_WOT   = 49152;                               // f16 [256][64]   32 KB
static constexpr size_t OFF_PART  = 81920;                               // f32 [2048]       8 KB
static constexpr size_t OFF_XPROJ = (size_t)1 << 20;                     // f32 [ROWS][128] 128 MB
static constexpr size_t OFF_FW    = OFF_XPROJ + (size_t)ROWS * HID * 4;  // f16 [ROWS][128]  64 MB
static constexpr size_t OFF_BW    = OFF_FW    + (size_t)ROWS * HID * 2;  // f16 [ROWS][128]  64 MB

__device__ inline void load8f(float* d, const float* s) {
    float4 x = ((const float4*)s)[0];
    float4 y = ((const float4*)s)[1];
    d[0]=x.x; d[1]=x.y; d[2]=x.z; d[3]=x.w;
    d[4]=y.x; d[5]=y.y; d[6]=y.z; d[7]=y.w;
}
__device__ inline void cvt8(_Float16* d, const float* s) {
#pragma unroll
    for (int i = 0; i < 8; ++i) d[i] = (_Float16)s[i];
}

// ---- Tensor Data Mover descriptor helpers (CDNA5 ISA) ----
// 2D tile: tile_d1 rows of tile_d0 elements; row stride stride0 (elements);
// dsz: 0=1B,1=2B,2=4B,3=8B. Groups 2/3 zeroed (2D tensor). 6-arg builtin form.
__device__ __forceinline__ void tdm_desc(u32x4& g0, i32x8& g1,
                                         const void* gaddr, unsigned lds_addr,
                                         unsigned tile_d0, unsigned tile_d1,
                                         unsigned tens_d0, unsigned tens_d1,
                                         unsigned stride0, unsigned dsz) {
    unsigned long long ga = (unsigned long long)(uintptr_t)gaddr;
    g0[0] = 1u;                                                   // count=1 (valid user D#)
    g0[1] = lds_addr;                                             // LDS byte address
    g0[2] = (unsigned)ga;                                         // global_addr[31:0]
    g0[3] = ((unsigned)(ga >> 32) & 0x01FFFFFFu) | (2u << 30);    // global_addr[56:32] | type=2
    g1[0] = (int)(dsz << 16);                                     // data_size; no mcast/iter/pad
    g1[1] = (int)((tens_d0 & 0xFFFFu) << 16);                     // tensor_dim0[15:0]
    g1[2] = (int)(((tens_d1 & 0xFFFFu) << 16) | ((tens_d0 >> 16) & 0xFFFFu));
    g1[3] = (int)(((tile_d0 & 0xFFFFu) << 16) | ((tens_d1 >> 16) & 0xFFFFu));
    g1[4] = (int)(tile_d1 & 0xFFFFu);                             // tile_dim1, tile_dim2=0
    g1[5] = (int)stride0;                                         // tensor_dim0_stride[31:0]
    g1[6] = 0;
    g1[7] = 0;
}

__device__ __forceinline__ void tdm_load_tile(const void* gaddr, unsigned lds_addr,
                                              unsigned tile_d0, unsigned tile_d1,
                                              unsigned stride0, unsigned dsz) {
    u32x4 g0; i32x8 g1;
    i32x4 z4 = {0, 0, 0, 0};
    i32x8 z8 = {0, 0, 0, 0, 0, 0, 0, 0};
    tdm_desc(g0, g1, gaddr, lds_addr, tile_d0, tile_d1, tile_d0, tile_d1, stride0, dsz);
    __builtin_amdgcn_tensor_load_to_lds(g0, g1, z4, z4, z8, 0);
}
__device__ __forceinline__ void tdm_store_tile(const void* gaddr, unsigned lds_addr,
                                               unsigned tile_d0, unsigned tile_d1,
                                               unsigned stride0, unsigned dsz) {
    u32x4 g0; i32x8 g1;
    i32x4 z4 = {0, 0, 0, 0};
    i32x8 z8 = {0, 0, 0, 0, 0, 0, 0, 0};
    tdm_desc(g0, g1, gaddr, lds_addr, tile_d0, tile_d1, tile_d0, tile_d1, stride0, dsz);
    __builtin_amdgcn_tensor_store_from_lds(g0, g1, z4, z4, z8, 0);
}

// ---- Kernel 1: weight conversion / transposition to f16 WMMA-friendly layouts ----
__global__ void prep_kernel(const float* __restrict__ Wf, const float* __restrict__ Wo,
                            _Float16* __restrict__ WxT, _Float16* __restrict__ WhT,
                            _Float16* __restrict__ WoT) {
    const int tid    = threadIdx.x + blockIdx.x * blockDim.x;
    const int stride = blockDim.x * gridDim.x;
    for (int i = tid; i < 64 * HID; i += stride) {            // WxT[k][o] = Wf[o][k]
        int k = i >> 7, o = i & 127;
        WxT[i] = (_Float16)Wf[o * (IN_DIM + HID) + k];
    }
    for (int i = tid; i < HID * HID; i += stride) {           // WhT[k][o] = Wf[o][64+k]
        int k = i >> 7, o = i & 127;
        WhT[i] = (_Float16)Wf[o * (IN_DIM + HID) + IN_DIM + k];
    }
    for (int i = tid; i < 2 * HID * IN_DIM; i += stride) {    // WoT[k][o] = Wo[o][k]
        int k = i >> 6, o = i & 63;
        WoT[i] = (_Float16)Wo[o * (2 * HID) + k];
    }
}

// ---- Kernel 2: xproj[R][o] = inps[R] @ Wx^T + bf  (fully parallel, K=64) ----
__global__ void __launch_bounds__(256) xproj_kernel(const float* __restrict__ inps,
                                                    const float* __restrict__ bf,
                                                    const _Float16* __restrict__ WxT,
                                                    float* __restrict__ xproj) {
    const int lane  = threadIdx.x & 31;
    const int wv    = threadIdx.x >> 5;      // N-tile 0..7
    const int mtile = blockIdx.x;            // 16 rows of (t,b)
    const int m  = lane & 15;
    const int hs = lane >> 4;
    const int ncol = wv * 16 + m;

    V16H b0, b1;
    {
        for (int s = 0; s < 16; ++s) {
            int kk = (s < 8) ? (hs * 8 + s) : (16 + hs * 8 + (s - 8));
            b0.h[s] = WxT[(size_t)kk * HID + ncol];
            b1.h[s] = WxT[(size_t)(32 + kk) * HID + ncol];
        }
    }
    const float* arow = inps + (size_t)(mtile * 16 + m) * IN_DIM;
    float t0[8], t1[8];
    V16H a0, a1;
    load8f(t0, arow + 0  + hs * 8);  load8f(t1, arow + 16 + hs * 8);
    cvt8(a0.h, t0);                  cvt8(a0.h + 8, t1);
    load8f(t0, arow + 32 + hs * 8);  load8f(t1, arow + 48 + hs * 8);
    cvt8(a1.h, t0);                  cvt8(a1.h + 8, t1);

    V8F c;
    const float bv = bf[ncol];
#pragma unroll
    for (int r = 0; r < 8; ++r) c.f[r] = bv;

    c.v = __builtin_amdgcn_wmma_f32_16x16x32_f16(false, a0.v, false, b0.v, (short)0, c.v, false, false);
    c.v = __builtin_amdgcn_wmma_f32_16x16x32_f16(false, a1.v, false, b1.v, (short)0, c.v, false, false);

#pragma unroll
    for (int r = 0; r < 8; ++r)
        xproj[(size_t)(mtile * 16 + r + 8 * hs) * HID + ncol] = c.f[r];
}

// ---- Kernel 3: bidirectional recurrence, TDM-pipelined ----
// 16 persistent WGs (2 dirs x 8 batch tiles). Per step:
//   * TDM async-loads next step's 16x128 f32 xproj tile into LDS (double buffered)
//   * C-init from current LDS xproj tile (ds loads), 4 WMMAs over h, tanh
//   * h tile written to LDS double buffer; TDM async-stores it to HBM
//   * one s_wait_tensorcnt 0 (wave 0) before the per-step barrier
__global__ void __launch_bounds__(256) rnn_kernel(const float* __restrict__ xproj,
                                                  const _Float16* __restrict__ WhT,
                                                  _Float16* __restrict__ fw,
                                                  _Float16* __restrict__ bw) {
    __shared__ _Float16 hbuf[2][16 * HID];   // 2 x 4 KB h tiles
    __shared__ float    xbuf[2][16 * HID];   // 2 x 8 KB xproj tiles
    const int lane  = threadIdx.x & 31;
    const int wv    = threadIdx.x >> 5;      // N-tile 0..7
    const int dir   = blockIdx.x >> 3;
    const int btile = blockIdx.x & 7;
    const int m  = lane & 15;
    const int hs = lane >> 4;
    const int ncol = wv * 16 + m;
    _Float16* __restrict__ hout = dir ? bw : fw;

    const unsigned lds_h0 = (unsigned)(uintptr_t)(const void*)&hbuf[0][0];
    const unsigned lds_h1 = (unsigned)(uintptr_t)(const void*)&hbuf[1][0];
    const unsigned lds_x0 = (unsigned)(uintptr_t)(const void*)&xbuf[0][0];
    const unsigned lds_x1 = (unsigned)(uintptr_t)(const void*)&xbuf[1][0];

    // Wh resident in VGPRs: 4 B-fragments (K = 0..127)
    V16H B[4];
#pragma unroll
    for (int kf = 0; kf < 4; ++kf) {
        for (int s = 0; s < 16; ++s) {
            int k = kf * 32 + ((s < 8) ? (hs * 8 + s) : (16 + hs * 8 + (s - 8)));
            B[kf].h[s] = WhT[(size_t)k * HID + ncol];
        }
    }
    for (int i = threadIdx.x; i < 16 * HID; i += 256) hbuf[0][i] = (_Float16)0.f;

    // Prologue: stage xproj tile for t=0
    const int t0eff = dir ? (SEQ_LEN - 1) : 0;
    if (wv == 0) {
        tdm_load_tile(xproj + ((size_t)t0eff * BATCH + btile * 16) * HID,
                      lds_x0, HID, 16, HID, /*dsz f32*/2u);
        __builtin_amdgcn_s_wait_tensorcnt(0);
    }
    __syncthreads();

    for (int t = 0; t < SEQ_LEN; ++t) {
        const int teff = dir ? (SEQ_LEN - 1 - t) : t;
        const size_t Rbase = (size_t)teff * BATCH + (size_t)btile * 16;
        const int pb = t & 1;

        // Kick off async load of next timestep's xproj tile into the other buffer
        if (wv == 0 && t + 1 < SEQ_LEN) {
            const int tn = dir ? (teff - 1) : (teff + 1);
            tdm_load_tile(xproj + ((size_t)tn * BATCH + btile * 16) * HID,
                          (pb ? lds_x0 : lds_x1), HID, 16, HID, 2u);
        }

        // C init from the staged xproj tile (LDS)
        const float* xb = &xbuf[pb][0];
        V8F c;
#pragma unroll
        for (int r = 0; r < 8; ++r)
            c.f[r] = xb[(r + 8 * hs) * HID + ncol];

#pragma unroll
        for (int kf = 0; kf < 4; ++kf) {
            V16H a;
            const _Float16* hr = &hbuf[pb][m * HID];
            a.q[0] = *(const float4*)(hr + kf * 32 + hs * 8);
            a.q[1] = *(const float4*)(hr + kf * 32 + 16 + hs * 8);
            c.v = __builtin_amdgcn_wmma_f32_16x16x32_f16(false, a.v, false, B[kf].v, (short)0, c.v, false, false);
        }

        // tanh, write next h tile into the other h buffer
        _Float16* hn = hbuf[pb ^ 1];
#pragma unroll
        for (int r = 0; r < 8; ++r) {
            float hv = tanhf(c.f[r]);
            hn[(r + 8 * hs) * HID + ncol] = (_Float16)hv;
        }

        // Complete outstanding TDM ops (next-x load + previous h store) before publishing
        if (wv == 0) __builtin_amdgcn_s_wait_tensorcnt(0);
        __syncthreads();

        // Async store of the finished h tile (16x128 f16 = flat 2048 elements)
        if (wv == 0) {
            tdm_store_tile(hout + Rbase * HID, (pb ? lds_h0 : lds_h1),
                           16 * HID, 1, 16 * HID, /*dsz f16*/1u);
        }
    }
    // S_ENDPGM performs an implicit wait-idle, draining the final TDM store.
}

// ---- Kernel 4: output projection (K=256) + log-softmax + per-block loss partial ----
__global__ void __launch_bounds__(256) loss_kernel(const _Float16* __restrict__ fw,
                                                   const _Float16* __restrict__ bw,
                                                   const _Float16* __restrict__ WoT,
                                                   const float* __restrict__ bo,
                                                   const float* __restrict__ targets,
                                                   float* __restrict__ partials) {
    __shared__ float ltile[8][16][64];
    __shared__ float wsum[8];
    const int lane = threadIdx.x & 31;
    const int wv   = threadIdx.x >> 5;
    const int m  = lane & 15;
    const int hs = lane >> 4;
    const size_t mtile = (size_t)blockIdx.x * 8 + wv;
    const size_t Rbase = mtile * 16;

    V16H A[8];
#pragma unroll
    for (int kf = 0; kf < 8; ++kf) {
        const _Float16* src = (kf < 4) ? fw : bw;
        const int kb = (kf & 3) * 32;
        const _Float16* p = src + (Rbase + m) * HID + kb;
        A[kf].q[0] = *(const float4*)(p + hs * 8);
        A[kf].q[1] = *(const float4*)(p + 16 + hs * 8);
    }

#pragma unroll
    for (int nt = 0; nt < 4; ++nt) {
        V8F c;
        const float bv = bo[nt * 16 + m];
#pragma unroll
        for (int r = 0; r < 8; ++r) c.f[r] = bv;
#pragma unroll
        for (int kf = 0; kf < 8; ++kf) {
            V16H b;
            for (int s = 0; s < 16; ++s) {
                int k = kf * 32 + ((s < 8) ? (hs * 8 + s) : (16 + hs * 8 + (s - 8)));
                b.h[s] = WoT[(size_t)k * 64 + nt * 16 + m];
            }
            c.v = __builtin_amdgcn_wmma_f32_16x16x32_f16(false, A[kf].v, false, b.v, (short)0, c.v, false, false);
        }
#pragma unroll
        for (int r = 0; r < 8; ++r)
            ltile[wv][r + 8 * hs][nt * 16 + m] = c.f[r];
    }
    __syncthreads();

    float lp = 0.f;
    if (lane < 16) {
        const int row = lane;
        const float* lr = ltile[wv][row];
        float mx = lr[0];
        for (int j = 1; j < 64; ++j) mx = fmaxf(mx, lr[j]);
        float s = 0.f;
        for (int j = 0; j < 64; ++j) s += expf(lr[j] - mx);
        const float logZ = mx + logf(s);
        const float* trow = targets + (Rbase + row) * IN_DIM;
        for (int j = 0; j < 64; ++j) lp += trow[j] * (lr[j] - logZ);
    }
    lp += __shfl_down(lp, 8, 32);
    lp += __shfl_down(lp, 4, 32);
    lp += __shfl_down(lp, 2, 32);
    lp += __shfl_down(lp, 1, 32);
    if (lane == 0) wsum[wv] = lp;
    __syncthreads();
    if (threadIdx.x == 0) {
        float s = 0.f;
#pragma unroll
        for (int i = 0; i < 8; ++i) s += wsum[i];
        partials[blockIdx.x] = s;
    }
}

// ---- Kernel 5: deterministic final reduction ----
__global__ void reduce_kernel(const float* __restrict__ partials, float* __restrict__ out) {
    __shared__ float sm[256];
    float s = 0.f;
    for (int i = threadIdx.x; i < 2048; i += 256) s += partials[i];
    sm[threadIdx.x] = s;
    __syncthreads();
    for (int off = 128; off > 0; off >>= 1) {
        if (threadIdx.x < off) sm[threadIdx.x] += sm[threadIdx.x + off];
        __syncthreads();
    }
    if (threadIdx.x == 0) out[0] = -sm[0] / (float)BATCH;
}

extern "C" void kernel_launch(void* const* d_in, const int* in_sizes, int n_in,
                              void* d_out, int out_size, void* d_ws, size_t ws_size,
                              hipStream_t stream) {
    const float* inps    = (const float*)d_in[0];
    const float* targets = (const float*)d_in[1];
    const float* Wf      = (const float*)d_in[2];
    const float* bf      = (const float*)d_in[3];
    const float* Wo      = (const float*)d_in[4];
    const float* bo      = (const float*)d_in[5];
    float* out = (float*)d_out;

    char* ws = (char*)d_ws;
    _Float16* WxT   = (_Float16*)(ws + OFF_WXT);
    _Float16* WhT   = (_Float16*)(ws + OFF_WHT);
    _Float16* WoT   = (_Float16*)(ws + OFF_WOT);
    float*    parts = (float*)   (ws + OFF_PART);
    float*    xproj = (float*)   (ws + OFF_XPROJ);
    _Float16* fwbuf = (_Float16*)(ws + OFF_FW);
    _Float16* bwbuf = (_Float16*)(ws + OFF_BW);

    prep_kernel  <<<64, 256, 0, stream>>>(Wf, Wo, WxT, WhT, WoT);
    xproj_kernel <<<ROWS / 16, 256, 0, stream>>>(inps, bf, WxT, xproj);
    rnn_kernel   <<<16, 256, 0, stream>>>(xproj, WhT, fwbuf, bwbuf);
    loss_kernel  <<<ROWS / 16 / 8, 256, 0, stream>>>(fwbuf, bwbuf, WoT, bo, targets, parts);
    reduce_kernel<<<1, 256, 0, stream>>>(parts, out);
}
